// pointnet2_cls_ssg_29111288332585
// MI455X (gfx1250) — compile-verified, matrix operand order checked by reading, not run on hardware
//
#include <hip/hip_runtime.h>
#include <hip/hip_bf16.h>

typedef __attribute__((ext_vector_type(16))) _Float16 v16h;
typedef __attribute__((ext_vector_type(8)))  _Float16 v8h;
typedef __attribute__((ext_vector_type(8)))  float    v8f;

// ---------------------------------------------------------------------------
// Utility kernels
// ---------------------------------------------------------------------------
__global__ void zero_f32_kernel(float* __restrict__ p, int n) {
    int t = blockIdx.x * blockDim.x + threadIdx.x;
    if (t < n) p[t] = 0.0f;
}

__global__ void f32_to_f16_kernel(const float* __restrict__ src, _Float16* __restrict__ dst, int n) {
    int t = blockIdx.x * blockDim.x + threadIdx.x;
    if (t < n) dst[t] = (_Float16)src[t];
}

// ---------------------------------------------------------------------------
// Farthest point sampling: one block per batch, 256 threads
// ---------------------------------------------------------------------------
__global__ void __launch_bounds__(256)
fps_kernel(const float* __restrict__ xyz, int N, int M,
           int* __restrict__ inds, float* __restrict__ new_xyz) {
    const int b = blockIdx.x;
    const int tid = threadIdx.x;
    const int NPB = N / 256;                 // 4 (N=1024) or 2 (N=512)
    const float* px = xyz + (size_t)b * N * 3;

    __shared__ float s_val[256];
    __shared__ int   s_idx[256];
    __shared__ float s_c[3];
    __shared__ int   s_cur;

    float mind[4];
#pragma unroll
    for (int j = 0; j < 4; ++j) mind[j] = 1e10f;

    if (tid == 0) { s_cur = 0; s_c[0] = px[0]; s_c[1] = px[1]; s_c[2] = px[2]; }
    __syncthreads();

    for (int it = 0; it < M; ++it) {
        const int cur = s_cur;
        const float cx = s_c[0], cy = s_c[1], cz = s_c[2];
        if (tid == 0) {
            inds[(size_t)b * M + it] = cur;
            new_xyz[((size_t)b * M + it) * 3 + 0] = cx;
            new_xyz[((size_t)b * M + it) * 3 + 1] = cy;
            new_xyz[((size_t)b * M + it) * 3 + 2] = cz;
        }
        float bestv = -1.0f; int besti = 0;
        for (int j = 0; j < NPB; ++j) {
            const int p = tid * NPB + j;
            const float dx = px[p * 3 + 0] - cx;
            const float dy = px[p * 3 + 1] - cy;
            const float dz = px[p * 3 + 2] - cz;
            const float d = dx * dx + dy * dy + dz * dz;
            const float m = fminf(mind[j], d);
            mind[j] = m;
            if (m > bestv) { bestv = m; besti = p; }
        }
        s_val[tid] = bestv; s_idx[tid] = besti;
        __syncthreads();
        for (int s = 128; s > 0; s >>= 1) {
            if (tid < s) {
                const float ov = s_val[tid + s]; const int oi = s_idx[tid + s];
                if (ov > s_val[tid] || (ov == s_val[tid] && oi < s_idx[tid])) {
                    s_val[tid] = ov; s_idx[tid] = oi;
                }
            }
            __syncthreads();
        }
        if (tid == 0) {
            const int nx = s_idx[0];
            s_cur = nx;
            s_c[0] = px[nx * 3 + 0]; s_c[1] = px[nx * 3 + 1]; s_c[2] = px[nx * 3 + 2];
        }
        __syncthreads();
    }
}

// ---------------------------------------------------------------------------
// Ball query: one thread per (b, m); scan in ascending index order
// ---------------------------------------------------------------------------
__global__ void ball_query_kernel(const float* __restrict__ xyz, const float* __restrict__ new_xyz,
                                  int B, int N, int M, int K, float r2, int* __restrict__ ginds) {
    const int t = blockIdx.x * blockDim.x + threadIdx.x;
    if (t >= B * M) return;
    const int b = t / M;
    const float* px = xyz + (size_t)b * N * 3;
    const float cx = new_xyz[(size_t)t * 3 + 0];
    const float cy = new_xyz[(size_t)t * 3 + 1];
    const float cz = new_xyz[(size_t)t * 3 + 2];
    int* g = ginds + (size_t)t * K;
    int cnt = 0, first = 0;
    for (int n = 0; n < N; ++n) {
        const float dx = px[n * 3 + 0] - cx;
        const float dy = px[n * 3 + 1] - cy;
        const float dz = px[n * 3 + 2] - cz;
        const float d = dx * dx + dy * dy + dz * dz;
        if (d <= r2) {
            if (cnt == 0) first = n;
            g[cnt] = n;
            if (++cnt >= K) break;
        }
    }
    for (int k = cnt; k < K; ++k) g[k] = first;
}

// ---------------------------------------------------------------------------
// Grouping: gather + center xyz, concat features, pad to Kpad, emit f16 rows
// ---------------------------------------------------------------------------
__global__ void group_kernel(const float* __restrict__ xyz, const float* __restrict__ feat,
                             const float* __restrict__ new_xyz, const int* __restrict__ ginds,
                             int B, int N, int M, int K, int Cf, int Kpad,
                             _Float16* __restrict__ A) {
    const int t = blockIdx.x * blockDim.x + threadIdx.x;
    if (t >= B * M * K) return;
    const int bm = t / K;
    const int b = bm / M;
    const int idx = ginds[t];
    const float* p = xyz + ((size_t)b * N + idx) * 3;
    const float* c = new_xyz + (size_t)bm * 3;
    _Float16* a = A + (size_t)t * Kpad;
    a[0] = (_Float16)(p[0] - c[0]);
    a[1] = (_Float16)(p[1] - c[1]);
    a[2] = (_Float16)(p[2] - c[2]);
    const float* f = feat + ((size_t)b * N + idx) * Cf;
    for (int i = 0; i < Cf; ++i) a[3 + i] = (_Float16)f[i];
    for (int i = 3 + Cf; i < Kpad; ++i) a[i] = (_Float16)0.0f;
}

// Group-all (SA3): no centering, one row per point
__global__ void groupall_kernel(const float* __restrict__ xyz, const float* __restrict__ feat,
                                int rows, int Cf, int Kpad, _Float16* __restrict__ A) {
    const int t = blockIdx.x * blockDim.x + threadIdx.x;
    if (t >= rows) return;
    const float* p = xyz + (size_t)t * 3;
    const float* f = feat + (size_t)t * Cf;
    _Float16* a = A + (size_t)t * Kpad;
    a[0] = (_Float16)p[0]; a[1] = (_Float16)p[1]; a[2] = (_Float16)p[2];
    for (int i = 0; i < Cf; ++i) a[3 + i] = (_Float16)f[i];
    for (int i = 3 + Cf; i < Kpad; ++i) a[i] = (_Float16)0.0f;
}

// ---------------------------------------------------------------------------
// Weight repack: f32 (Cin, Cout) -> f16 B-fragment tiles (lane-major, 16 halfs
// per lane, one 32B contiguous load per lane per 32x16 K-tile)
// ---------------------------------------------------------------------------
__global__ void pack_w_kernel(const float* __restrict__ W, int Cin, int CoutAct,
                              int KT, int NT, _Float16* __restrict__ Bp) {
    const int t = blockIdx.x * blockDim.x + threadIdx.x;
    const int total = KT * NT * 32;
    if (t >= total) return;
    const int lane = t & 31;
    const int tile = t >> 5;
    const int nt = tile % NT;
    const int kt = tile / NT;
    const int n = nt * 16 + (lane & 15);
    const int half = lane >> 4;
    _Float16* o = Bp + (size_t)t * 16;
#pragma unroll
    for (int i = 0; i < 8; ++i) {
        const int k0 = kt * 32 + half * 8 + i;
        const int k1 = k0 + 16;
        o[i]     = (k0 < Cin && n < CoutAct) ? (_Float16)W[(size_t)k0 * CoutAct + n] : (_Float16)0.0f;
        o[i + 8] = (k1 < Cin && n < CoutAct) ? (_Float16)W[(size_t)k1 * CoutAct + n] : (_Float16)0.0f;
    }
}

// ---------------------------------------------------------------------------
// WMMA GEMM, register-blocked 2x2: each wave computes a 32x32 output block
// (4 x v_wmma_f32_16x16x32_f16 per 32-wide K step, reusing 2 A + 2 B frags).
// Requires rows % 32 == 0 and CoutPad % 32 == 0 (true for all layers here).
// Optionally accumulates per-channel sum / sum-of-squares for batch-norm.
// ---------------------------------------------------------------------------
__global__ void __launch_bounds__(256)
gemm_wmma_kernel(const _Float16* __restrict__ A, const _Float16* __restrict__ Bp,
                 float* __restrict__ Y, int rows, int Kpad, int CoutPad,
                 float* __restrict__ sums, float* __restrict__ sumsq, int doStats) {
    const int lane = threadIdx.x & 31;
    const int wave = blockIdx.x * (blockDim.x >> 5) + (threadIdx.x >> 5);
    const int NT = CoutPad >> 4;
    const int NT2 = NT >> 1;
    const int RT2 = rows >> 5;
    if (wave >= RT2 * NT2) return;          // uniform per wave: EXEC all-ones inside
    const int rt2 = wave / NT2;
    const int nt2 = wave - rt2 * NT2;
    const int KT = Kpad >> 5;

    // A fragment: lane holds row (lane&15); halfs [base..base+7] and [base+16..]
    const int arow0 = rt2 * 32 + (lane & 15);
    const _Float16* pa0 = A + (size_t)arow0 * Kpad + ((lane >> 4) << 3);
    const _Float16* pa1 = pa0 + (size_t)16 * Kpad;
    const _Float16* pb0 = Bp + ((size_t)(nt2 * 2) * 32 + lane) * 16;
    const size_t pbStep = (size_t)NT * 512;  // one K-tile worth of packed B

    v8f c00 = {}, c01 = {}, c10 = {}, c11 = {};
    for (int kt = 0; kt < KT; ++kt) {
        __builtin_prefetch(pa0 + 32, 0, 1);  // global_prefetch for next K-tile of A
        __builtin_prefetch(pa1 + 32, 0, 1);
        const v8h lo0 = *(const v8h*)(pa0);
        const v8h hi0 = *(const v8h*)(pa0 + 16);
        const v8h lo1 = *(const v8h*)(pa1);
        const v8h hi1 = *(const v8h*)(pa1 + 16);
        v16h a0, a1;
#pragma unroll
        for (int i = 0; i < 8; ++i) {
            a0[i] = lo0[i]; a0[i + 8] = hi0[i];
            a1[i] = lo1[i]; a1[i + 8] = hi1[i];
        }
        const v16h b0 = *(const v16h*)(pb0);
        const v16h b1 = *(const v16h*)(pb0 + 512);
        c00 = __builtin_amdgcn_wmma_f32_16x16x32_f16(false, a0, false, b0, (short)0, c00, false, false);
        c01 = __builtin_amdgcn_wmma_f32_16x16x32_f16(false, a0, false, b1, (short)0, c01, false, false);
        c10 = __builtin_amdgcn_wmma_f32_16x16x32_f16(false, a1, false, b0, (short)0, c10, false, false);
        c11 = __builtin_amdgcn_wmma_f32_16x16x32_f16(false, a1, false, b1, (short)0, c11, false, false);
        pa0 += 32;
        pa1 += 32;
        pb0 += pbStep;
    }

    // C/D layout: VGPR r, lane l -> (M = r + 8*(l>>4), N = l&15)
    const int col0 = nt2 * 32 + (lane & 15);
    const int col1 = col0 + 16;
    const int rb0 = rt2 * 32 + ((lane >> 4) << 3);
    const int rb1 = rb0 + 16;
    float s0 = 0.0f, q0 = 0.0f, s1 = 0.0f, q1 = 0.0f;
#pragma unroll
    for (int r = 0; r < 8; ++r) {
        const float v00 = c00[r], v01 = c01[r], v10 = c10[r], v11 = c11[r];
        Y[(size_t)(rb0 + r) * CoutPad + col0] = v00;
        Y[(size_t)(rb0 + r) * CoutPad + col1] = v01;
        Y[(size_t)(rb1 + r) * CoutPad + col0] = v10;
        Y[(size_t)(rb1 + r) * CoutPad + col1] = v11;
        s0 += v00 + v10; q0 += v00 * v00 + v10 * v10;
        s1 += v01 + v11; q1 += v01 * v01 + v11 * v11;
    }
    if (doStats) {
        s0 += __shfl_xor(s0, 16);
        q0 += __shfl_xor(q0, 16);
        s1 += __shfl_xor(s1, 16);
        q1 += __shfl_xor(q1, 16);
        if (lane < 16) {
            atomicAdd(&sums[col0], s0);
            atomicAdd(&sumsq[col0], q0);
            atomicAdd(&sums[col1], s1);
            atomicAdd(&sumsq[col1], q1);
        }
    }
}

// ---------------------------------------------------------------------------
// BatchNorm (training-mode forward) helpers
// ---------------------------------------------------------------------------
__global__ void bn_stats_kernel(const float* __restrict__ sums, const float* __restrict__ sumsq,
                                const float* __restrict__ gamma, const float* __restrict__ beta,
                                int C, float invCnt, float* __restrict__ scale, float* __restrict__ shift) {
    const int c = blockIdx.x * blockDim.x + threadIdx.x;
    if (c >= C) return;
    const float mean = sums[c] * invCnt;
    const float var = sumsq[c] * invCnt - mean * mean;
    const float sc = gamma[c] * rsqrtf(var + 1e-5f);
    scale[c] = sc;
    shift[c] = beta[c] - mean * sc;
}

__global__ void bn_relu_f16_kernel(const float* __restrict__ Y, const float* __restrict__ scale,
                                   const float* __restrict__ shift, size_t total, int C,
                                   _Float16* __restrict__ Anext) {
    const size_t t = (size_t)blockIdx.x * blockDim.x + threadIdx.x;
    if (t >= total) return;
    const int c = (int)(t % C);
    const float v = Y[t] * scale[c] + shift[c];
    Anext[t] = (_Float16)fmaxf(v, 0.0f);
}

__global__ void bn_relu_inplace_kernel(float* __restrict__ Y, const float* __restrict__ scale,
                                       const float* __restrict__ shift, size_t total, int C) {
    const size_t t = (size_t)blockIdx.x * blockDim.x + threadIdx.x;
    if (t >= total) return;
    const int c = (int)(t % C);
    Y[t] = fmaxf(Y[t] * scale[c] + shift[c], 0.0f);
}

// Max-pool over group dimension K: in (BM*K, C) -> out (BM, C)
__global__ void maxpool_kernel(const float* __restrict__ Y, int BM, int K, int C,
                               float* __restrict__ out) {
    const int t = blockIdx.x * blockDim.x + threadIdx.x;
    if (t >= BM * C) return;
    const int c = t % C;
    const int bm = t / C;
    const float* p = Y + (size_t)bm * K * C + c;
    float m = p[0];
    for (int k = 1; k < K; ++k) m = fmaxf(m, p[(size_t)k * C]);
    out[t] = m;
}

// Final classifier: add bias, strip column padding (64 -> 40)
__global__ void bias_out_kernel(const float* __restrict__ Y, const float* __restrict__ bias,
                                float* __restrict__ out, int B, int Cout, int CoutPad) {
    const int t = blockIdx.x * blockDim.x + threadIdx.x;
    if (t >= B * Cout) return;
    const int b = t / Cout;
    const int c = t % Cout;
    out[t] = Y[(size_t)b * CoutPad + c] + bias[c];
}

// ---------------------------------------------------------------------------
// Host orchestration
// ---------------------------------------------------------------------------
extern "C" void kernel_launch(void* const* d_in, const int* in_sizes, int n_in,
                              void* d_out, int out_size, void* d_ws, size_t ws_size,
                              hipStream_t stream) {
    (void)in_sizes; (void)n_in; (void)out_size; (void)ws_size;

    const int B = 32;
    const int R1 = B * 512 * 32;     // 524288 rows SA1
    const int R2 = B * 128 * 64;     // 262144 rows SA2
    const int R3 = B * 128;          // 4096   rows SA3
    const int RF = B;                // 32     rows FC

    // ---- inputs --------------------------------------------------------
    const float* xyz = (const float*)d_in[0];
    const float* pts = (const float*)d_in[1];
    const float* W_[12]; const float* G_[12]; const float* Bt_[12];
    for (int l = 0; l < 3; ++l) { W_[l]     = (const float*)d_in[2 + 3 * l];
                                  G_[l]     = (const float*)d_in[3 + 3 * l];
                                  Bt_[l]    = (const float*)d_in[4 + 3 * l]; }
    for (int l = 0; l < 3; ++l) { W_[3 + l] = (const float*)d_in[11 + 3 * l];
                                  G_[3 + l] = (const float*)d_in[12 + 3 * l];
                                  Bt_[3 + l]= (const float*)d_in[13 + 3 * l]; }
    for (int l = 0; l < 3; ++l) { W_[6 + l] = (const float*)d_in[20 + 3 * l];
                                  G_[6 + l] = (const float*)d_in[21 + 3 * l];
                                  Bt_[6 + l]= (const float*)d_in[22 + 3 * l]; }
    W_[9]  = (const float*)d_in[29]; G_[9]  = (const float*)d_in[30]; Bt_[9]  = (const float*)d_in[31];
    W_[10] = (const float*)d_in[32]; G_[10] = (const float*)d_in[33]; Bt_[10] = (const float*)d_in[34];
    W_[11] = (const float*)d_in[35];
    const float* clsB = (const float*)d_in[36];

    // ---- workspace carve (deterministic every call) --------------------
    char* base = (char*)d_ws;
    size_t off = 0;
    auto alloc = [&](size_t bytes) -> void* {
        void* p = base + off;
        off = (off + bytes + 255) & ~(size_t)255;
        return p;
    };

    float*     Yb    = (float*)    alloc((size_t)67108864 * sizeof(float));     // max rows*Cout
    _Float16*  A0    = (_Float16*) alloc((size_t)R2 * 160 * sizeof(_Float16));  // 41.9M halfs
    _Float16*  A1    = (_Float16*) alloc((size_t)R1 * 64  * sizeof(_Float16));  // 33.5M halfs
    float*     feat1 = (float*)    alloc((size_t)B * 512 * 128 * sizeof(float));
    float*     feat2 = (float*)    alloc((size_t)B * 128 * 256 * sizeof(float));
    float*     feat3 = (float*)    alloc((size_t)B * 1024 * sizeof(float));
    float*     nx1   = (float*)    alloc((size_t)B * 512 * 3 * sizeof(float));
    float*     nx2   = (float*)    alloc((size_t)B * 128 * 3 * sizeof(float));
    int*       inds1 = (int*)      alloc((size_t)B * 512 * sizeof(int));
    int*       inds2 = (int*)      alloc((size_t)B * 128 * sizeof(int));
    int*       gi1   = (int*)      alloc((size_t)B * 512 * 32 * sizeof(int));
    int*       gi2   = (int*)      alloc((size_t)B * 128 * 64 * sizeof(int));
    float*     sums  = (float*)    alloc(1024 * sizeof(float));
    float*     sumsq = (float*)    alloc(1024 * sizeof(float));
    float*     scl   = (float*)    alloc(1024 * sizeof(float));
    float*     shf   = (float*)    alloc(1024 * sizeof(float));

    auto cdiv = [](size_t a, size_t b) { return (int)((a + b - 1) / b); };

    // One GEMM (+optional BN+ReLU) layer
    auto gemm_layer = [&](const _Float16* Ain, const float* W, int Cin, int Kpad,
                          int CoutAct, int CoutPad, int rows,
                          const float* gamma, const float* beta,
                          _Float16* Anext /* null => in-place f32 (or skip if no BN) */) {
        const int KT = Kpad / 32, NT = CoutPad / 16;
        _Float16* Bp = (_Float16*)alloc((size_t)Kpad * CoutPad * sizeof(_Float16));
        const int packTot = KT * NT * 32;
        pack_w_kernel<<<cdiv(packTot, 256), 256, 0, stream>>>(W, Cin, CoutAct, KT, NT, Bp);

        const bool stats = (gamma != nullptr);
        if (stats) {
            zero_f32_kernel<<<cdiv(CoutPad, 256), 256, 0, stream>>>(sums, CoutPad);
            zero_f32_kernel<<<cdiv(CoutPad, 256), 256, 0, stream>>>(sumsq, CoutPad);
        }
        const int tiles2 = (rows / 32) * (NT / 2);   // 2x2-blocked waves
        gemm_wmma_kernel<<<cdiv(tiles2, 8), 256, 0, stream>>>(Ain, Bp, Yb, rows, Kpad, CoutPad,
                                                              sums, sumsq, stats ? 1 : 0);
        if (stats) {
            bn_stats_kernel<<<cdiv(CoutPad, 256), 256, 0, stream>>>(sums, sumsq, gamma, beta,
                                                                    CoutPad, 1.0f / (float)rows, scl, shf);
            const size_t total = (size_t)rows * CoutPad;
            if (Anext)
                bn_relu_f16_kernel<<<cdiv(total, 256), 256, 0, stream>>>(Yb, scl, shf, total, CoutPad, Anext);
            else
                bn_relu_inplace_kernel<<<cdiv(total, 256), 256, 0, stream>>>(Yb, scl, shf, total, CoutPad);
        }
    };

    // ---- SA1: N=1024 -> M=512, r=0.2, K=32, MLP 6->64->64->128 ---------
    fps_kernel<<<B, 256, 0, stream>>>(xyz, 1024, 512, inds1, nx1);
    ball_query_kernel<<<cdiv(B * 512, 256), 256, 0, stream>>>(xyz, nx1, B, 1024, 512, 32, 0.04f, gi1);
    group_kernel<<<cdiv(R1, 256), 256, 0, stream>>>(xyz, pts, nx1, gi1, B, 1024, 512, 32, 3, 32, A0);
    gemm_layer(A0, W_[0],   6,  32,  64,  64, R1, G_[0], Bt_[0], A1);
    gemm_layer(A1, W_[1],  64,  64,  64,  64, R1, G_[1], Bt_[1], A0);
    gemm_layer(A0, W_[2],  64,  64, 128, 128, R1, G_[2], Bt_[2], nullptr);
    maxpool_kernel<<<cdiv((size_t)B * 512 * 128, 256), 256, 0, stream>>>(Yb, B * 512, 32, 128, feat1);

    // ---- SA2: N=512 -> M=128, r=0.4, K=64, MLP 131->128->128->256 ------
    fps_kernel<<<B, 256, 0, stream>>>(nx1, 512, 128, inds2, nx2);
    ball_query_kernel<<<cdiv(B * 128, 256), 256, 0, stream>>>(nx1, nx2, B, 512, 128, 64, 0.16f, gi2);
    group_kernel<<<cdiv(R2, 256), 256, 0, stream>>>(nx1, feat1, nx2, gi2, B, 512, 128, 64, 128, 160, A0);
    gemm_layer(A0, W_[3], 131, 160, 128, 128, R2, G_[3], Bt_[3], A1);
    gemm_layer(A1, W_[4], 128, 128, 128, 128, R2, G_[4], Bt_[4], A0);
    gemm_layer(A0, W_[5], 128, 128, 256, 256, R2, G_[5], Bt_[5], nullptr);
    maxpool_kernel<<<cdiv((size_t)B * 128 * 256, 256), 256, 0, stream>>>(Yb, B * 128, 64, 256, feat2);

    // ---- SA3 (group-all): rows=4096, MLP 259->256->512->1024 -----------
    groupall_kernel<<<cdiv(R3, 256), 256, 0, stream>>>(nx2, feat2, R3, 256, 288, A0);
    gemm_layer(A0, W_[6], 259, 288,  256,  256, R3, G_[6], Bt_[6], A1);
    gemm_layer(A1, W_[7], 256, 256,  512,  512, R3, G_[7], Bt_[7], A0);
    gemm_layer(A0, W_[8], 512, 512, 1024, 1024, R3, G_[8], Bt_[8], nullptr);
    maxpool_kernel<<<cdiv((size_t)B * 1024, 256), 256, 0, stream>>>(Yb, B, 128, 1024, feat3);

    // ---- FC head: 1024->512->256->40(+bias) ----------------------------
    f32_to_f16_kernel<<<cdiv(B * 1024, 256), 256, 0, stream>>>(feat3, A0, B * 1024);
    gemm_layer(A0, W_[9],  1024, 1024, 512, 512, RF, G_[9],  Bt_[9],  A1);
    gemm_layer(A1, W_[10],  512,  512, 256, 256, RF, G_[10], Bt_[10], A0);
    gemm_layer(A0, W_[11],  256,  256,  40,  64, RF, nullptr, nullptr, nullptr);
    bias_out_kernel<<<cdiv(B * 40, 256), 256, 0, stream>>>(Yb, clsB, (float*)d_out, B, 40, 64);
}
